// CrossLayerBlock_64141041598887
// MI455X (gfx1250) — matrix operand order repaired and checked by reading, not compiled, not run on hardware
//
#include <hip/hip_runtime.h>
#include <hip/hip_bf16.h>

// ---------------- problem constants ----------------
#define BB   4
#define TT   1024
#define DD   1024
#define HH   16
#define EE   8
#define KTOP 2
#define HSZ  64
#define DFFC 4096
#define NT   4096          // B*T tokens
#define CAPM 1024          // NT*K/E

typedef __attribute__((ext_vector_type(16))) _Float16 v16h;
typedef __attribute__((ext_vector_type(8)))  _Float16 v8h;
typedef __attribute__((ext_vector_type(8)))  float    v8f;

// ---------------- WMMA fragment loaders (contiguous k order, shared by A and B, wave32) ------
// A fragment 16x32 f16: lane l(0..15)=row M, khi=l>>4; halves i =
// K khi*16+i. Contiguous -> 2x b128.
__device__ __forceinline__ v16h frag_a(const _Float16* p, int khi) {
  v8h lo = *(const v8h*)(p + khi * 16);
  v8h hi = *(const v8h*)(p + khi * 16 + 8);
  v16h r;
#pragma unroll
  for (int i = 0; i < 8; ++i) { r[i] = lo[i]; r[i + 8] = hi[i]; }
  return r;
}
// B fragment 32x16 f16 from transposed operand Bt[n][k]: lane n, halves i =
// K khi*16+i. Contiguous -> 2x b128.
__device__ __forceinline__ v16h frag_b(const _Float16* p, int khi) {
  v8h lo = *(const v8h*)(p + khi * 16);
  v8h hi = *(const v8h*)(p + khi * 16 + 8);
  v16h r;
#pragma unroll
  for (int i = 0; i < 8; ++i) { r[i] = lo[i]; r[i + 8] = hi[i]; }
  return r;
}

// ---------------- generic NT WMMA GEMM ----------------
// C[M,N] = epi( A[M,Kd]f16 @ Bt[N,Kd]f16 )
// block = 256 thr = 8 waves; block tile 64x128; wave tile 32x32:
// per 32-K step: 2 A frags + 2 B frags -> 4 wmma (16 f16-FLOP/byte at the
// load pipe, 2x the 16x32 tile). B frags shared by 2 M-slots via WGP$.
template <bool GATHER, bool RELU, bool HASB, bool HASR, bool W32, bool W16, bool SCAT>
__global__ __launch_bounds__(256) void gemm_nt(
    const _Float16* __restrict__ A, int lda, long long aB,
    const _Float16* __restrict__ Bt, int ldb, long long bB,
    const float* __restrict__ bias, long long biasB,
    const float* __restrict__ res, int ldr,
    float* __restrict__ C32, long long cB,
    _Float16* __restrict__ C16, long long c16B, int ldc,
    const int* __restrict__ tok, long long tokB,
    const float* __restrict__ gsel, long long gselB,
    int Kd, int nTokens) {
  int z = blockIdx.z;
  A += (long long)z * aB;
  Bt += (long long)z * bB;
  if (HASB) bias += (long long)z * biasB;
  if (W32 || SCAT) C32 += (long long)z * cB;
  if (W16) C16 += (long long)z * c16B;
  if (GATHER || SCAT) tok += (long long)z * tokB;
  if (SCAT) gsel += (long long)z * gselB;

  int wid = threadIdx.x >> 5, lane = threadIdx.x & 31;
  int l15 = lane & 15, khi = (lane >> 4) & 1;
  int mBase = blockIdx.y * 64 + (wid & 1) * 32;   // 2 M-slots of 32
  int nBase = blockIdx.x * 128 + (wid >> 1) * 32; // 4 N-slots of 32

  long long aIdx0 = 0, aIdx1 = 0;
  bool av0 = true, av1 = true;
  if (GATHER) {
    int tr0 = tok[mBase + l15];
    int tr1 = tok[mBase + 16 + l15];
    av0 = tr0 < nTokens;
    av1 = tr1 < nTokens;
    aIdx0 = av0 ? (long long)tr0 * lda : 0;
    aIdx1 = av1 ? (long long)tr1 * lda : 0;
  } else {
    aIdx0 = (long long)(mBase + l15) * lda;
    aIdx1 = (long long)(mBase + 16 + l15) * lda;
  }
  const _Float16* bRow0 = Bt + (long long)(nBase + l15) * ldb;
  const _Float16* bRow1 = Bt + (long long)(nBase + 16 + l15) * ldb;

  v8f acc00 = {}, acc01 = {}, acc10 = {}, acc11 = {};
  for (int k = 0; k < Kd; k += 32) {
    if (k + 64 < Kd) {  // prefetch 2 K-steps ahead -> global_prefetch_b8
      __builtin_prefetch(A + aIdx0 + k + 64, 0, 1);
      __builtin_prefetch(A + aIdx1 + k + 64, 0, 1);
      __builtin_prefetch(bRow0 + k + 64, 0, 1);
      __builtin_prefetch(bRow1 + k + 64, 0, 1);
    }
    v16h af0 = {}, af1 = {};
    if (!GATHER || av0) af0 = frag_a(A + aIdx0 + k, khi);
    if (!GATHER || av1) af1 = frag_a(A + aIdx1 + k, khi);
    v16h bf0 = frag_b(bRow0 + k, khi);
    v16h bf1 = frag_b(bRow1 + k, khi);
    acc00 = __builtin_amdgcn_wmma_f32_16x16x32_f16(false, af0, false, bf0, (short)0, acc00, false, false);
    acc01 = __builtin_amdgcn_wmma_f32_16x16x32_f16(false, af0, false, bf1, (short)0, acc01, false, false);
    acc10 = __builtin_amdgcn_wmma_f32_16x16x32_f16(false, af1, false, bf0, (short)0, acc10, false, false);
    acc11 = __builtin_amdgcn_wmma_f32_16x16x32_f16(false, af1, false, bf1, (short)0, acc11, false, false);
  }

  // Epilogue: C layout -> row m = khi*8+r (per 16-row subtile), col = l15.
#pragma unroll
  for (int mt = 0; mt < 2; ++mt) {
    v8f* a0 = (mt == 0) ? &acc00 : &acc10;
    v8f* a1 = (mt == 0) ? &acc01 : &acc11;
#pragma unroll
    for (int r = 0; r < 8; ++r) {
      int row = mBase + mt * 16 + khi * 8 + r;
      float v0 = (*a0)[r], v1 = (*a1)[r];
      int c0 = nBase + l15, c1 = nBase + 16 + l15;
      if (HASB) { v0 += bias[c0]; v1 += bias[c1]; }
      if (RELU) { v0 = fmaxf(v0, 0.f); v1 = fmaxf(v1, 0.f); }
      if (HASR) {
        v0 += res[(long long)row * ldr + c0];
        v1 += res[(long long)row * ldr + c1];
      }
      if (W32) {
        C32[(long long)row * ldc + c0] = v0;
        C32[(long long)row * ldc + c1] = v1;
      }
      if (W16) {
        C16[(long long)row * ldc + c0] = (_Float16)v0;
        C16[(long long)row * ldc + c1] = (_Float16)v1;
      }
      if (SCAT) {
        int tr = tok[row];
        if (tr < nTokens) {
          float g = gsel[row];
          atomicAdd(&C32[(long long)tr * ldc + c0], v0 * g);
          atomicAdd(&C32[(long long)tr * ldc + c1], v1 * g);
        }
      }
    }
  }
}

// ---------------- flash attention (causal, online softmax) ----------------
// grid (T/64, B*H), block 128 = 4 waves; each wave owns 16 query rows.
__global__ __launch_bounds__(128) void flash_attn(
    const _Float16* __restrict__ qf, const _Float16* __restrict__ kf,
    const _Float16* __restrict__ vt, _Float16* __restrict__ attn16) {
  int bh = blockIdx.y;
  int b = bh / HH, h = bh % HH;
  int qBase = blockIdx.x * 64;
  int wid = threadIdx.x >> 5, lane = threadIdx.x & 31;
  int l15 = lane & 15, khi = (lane >> 4) & 1;
  int myQ = qBase + wid * 16;

  const _Float16* qrow = qf + ((long long)(b * TT + myQ + l15)) * DD + h * HSZ;
  v16h aq0 = frag_a(qrow, khi);       // hs 0..31
  v16h aq1 = frag_a(qrow + 32, khi);  // hs 32..63

  v8f o0 = {}, o1 = {}, o2 = {}, o3 = {};
  float mrow[8], lrow[8];
#pragma unroll
  for (int r = 0; r < 8; ++r) { mrow[r] = -1e30f; lrow[r] = 0.f; }

  __shared__ __align__(16) _Float16 pT[4][16 * 32];
  _Float16* pt = pT[wid];
  const float scale = 0.03125f;  // D^-0.5 (D=1024, per reference)

  for (int kt = 0; kt < qBase + 64; kt += 32) {
    if (kt > myQ + 15) continue;  // wave-uniform causal skip
    const _Float16* krow0 = kf + ((long long)(b * TT + kt + l15)) * DD + h * HSZ;
    const _Float16* krow1 = kf + ((long long)(b * TT + kt + 16 + l15)) * DD + h * HSZ;
    v8f s0 = {}, s1 = {};
    s0 = __builtin_amdgcn_wmma_f32_16x16x32_f16(false, aq0, false, frag_b(krow0, khi), (short)0, s0, false, false);
    s0 = __builtin_amdgcn_wmma_f32_16x16x32_f16(false, aq1, false, frag_b(krow0 + 32, khi), (short)0, s0, false, false);
    s1 = __builtin_amdgcn_wmma_f32_16x16x32_f16(false, aq0, false, frag_b(krow1, khi), (short)0, s1, false, false);
    s1 = __builtin_amdgcn_wmma_f32_16x16x32_f16(false, aq1, false, frag_b(krow1 + 32, khi), (short)0, s1, false, false);

    int col0 = kt + l15, col1 = kt + 16 + l15;
    float pv0[8], pv1[8], alpha[8];
#pragma unroll
    for (int r = 0; r < 8; ++r) {
      int qr = myQ + khi * 8 + r;
      float v0 = (col0 <= qr) ? s0[r] * scale : -1e30f;
      float v1 = (col1 <= qr) ? s1[r] * scale : -1e30f;
      float mx = fmaxf(v0, v1);
#pragma unroll
      for (int off = 8; off; off >>= 1) mx = fmaxf(mx, __shfl_xor(mx, off, 16));
      float mnew = fmaxf(mrow[r], mx);
      float a = __expf(mrow[r] - mnew);
      float p0 = __expf(v0 - mnew);
      float p1 = __expf(v1 - mnew);
      float sum = p0 + p1;
#pragma unroll
      for (int off = 8; off; off >>= 1) sum += __shfl_xor(sum, off, 16);
      lrow[r] = lrow[r] * a + sum;
      mrow[r] = mnew;
      alpha[r] = a; pv0[r] = p0; pv1[r] = p1;
    }
#pragma unroll
    for (int r = 0; r < 8; ++r) {
      o0[r] *= alpha[r]; o1[r] *= alpha[r]; o2[r] *= alpha[r]; o3[r] *= alpha[r];
    }
    // P (C layout) -> LDS row-major 16x32 -> A fragment (cross-lane transpose)
#pragma unroll
    for (int r = 0; r < 8; ++r) {
      int rw = khi * 8 + r;
      pt[rw * 32 + l15] = (_Float16)pv0[r];
      pt[rw * 32 + 16 + l15] = (_Float16)pv1[r];
    }
    __asm__ volatile("s_wait_dscnt 0" ::: "memory");
    v16h ap = frag_a(pt + l15 * 32, khi);
    const _Float16* vb = vt + ((long long)bh * HSZ) * TT + kt;
    o0 = __builtin_amdgcn_wmma_f32_16x16x32_f16(false, ap, false, frag_b(vb + (long long)(0 + l15) * TT, khi), (short)0, o0, false, false);
    o1 = __builtin_amdgcn_wmma_f32_16x16x32_f16(false, ap, false, frag_b(vb + (long long)(16 + l15) * TT, khi), (short)0, o1, false, false);
    o2 = __builtin_amdgcn_wmma_f32_16x16x32_f16(false, ap, false, frag_b(vb + (long long)(32 + l15) * TT, khi), (short)0, o2, false, false);
    o3 = __builtin_amdgcn_wmma_f32_16x16x32_f16(false, ap, false, frag_b(vb + (long long)(48 + l15) * TT, khi), (short)0, o3, false, false);
  }

#pragma unroll
  for (int r = 0; r < 8; ++r) {
    int m = khi * 8 + r;
    float inv = 1.f / lrow[r];
    _Float16* dst = attn16 + ((long long)(b * TT + myQ + m)) * DD + h * HSZ + l15;
    dst[0]  = (_Float16)(o0[r] * inv);
    dst[16] = (_Float16)(o1[r] * inv);
    dst[32] = (_Float16)(o2[r] * inv);
    dst[48] = (_Float16)(o3[r] * inv);
  }
}

// ---------------- LayerNorm (one block per row, D=1024) ----------------
__global__ __launch_bounds__(256) void ln_kernel(
    const float* __restrict__ x, const float* __restrict__ g,
    const float* __restrict__ bta, float* __restrict__ y32,
    _Float16* __restrict__ y16) {
  int row = blockIdx.x, tid = threadIdx.x;
  const float* xr = x + (long long)row * DD;
  float vals[4], s = 0.f, s2 = 0.f;
#pragma unroll
  for (int i = 0; i < 4; ++i) {
    float v = xr[tid + i * 256];
    vals[i] = v; s += v; s2 += v * v;
  }
  __shared__ float sbuf[256];
  sbuf[tid] = s; __syncthreads();
  for (int off = 128; off; off >>= 1) { if (tid < off) sbuf[tid] += sbuf[tid + off]; __syncthreads(); }
  float mean = sbuf[0] * (1.f / DD);
  __syncthreads();
  sbuf[tid] = s2; __syncthreads();
  for (int off = 128; off; off >>= 1) { if (tid < off) sbuf[tid] += sbuf[tid + off]; __syncthreads(); }
  float var = sbuf[0] * (1.f / DD) - mean * mean;
  float inv = rsqrtf(var + 1e-5f);
#pragma unroll
  for (int i = 0; i < 4; ++i) {
    int c = tid + i * 256;
    float o = (vals[i] - mean) * inv * g[c] + bta[c];
    y32[(long long)row * DD + c] = o;
    y16[(long long)row * DD + c] = (_Float16)o;
  }
}

// ---------------- weight transpose+convert f32[K,N] -> f16[N,K] ------------
__global__ void transpose_cvt(const float* __restrict__ src, _Float16* __restrict__ dst,
                              int Kd, int Nd, long long srcB, long long dstB) {
  src += (long long)blockIdx.z * srcB;
  dst += (long long)blockIdx.z * dstB;
  long long i = (long long)blockIdx.x * blockDim.x + threadIdx.x;
  if (i >= (long long)Kd * Nd) return;
  int n = (int)(i % Nd);
  long long k = i / Nd;
  dst[(long long)n * Kd + k] = (_Float16)src[i];
}

// ---------------- V repack: [B,T,H*HS]f16 -> [B*H, HS, T]f16 ----------------
__global__ void pack_vt(const _Float16* __restrict__ vf, _Float16* __restrict__ vtb) {
  long long i = (long long)blockIdx.x * blockDim.x + threadIdx.x;
  if (i >= (long long)NT * DD) return;
  int t = (int)(i % TT);
  long long r = i / TT;
  int hs = (int)(r % HSZ);
  int bh = (int)(r / HSZ);
  int b = bh / HH, h = bh % HH;
  vtb[i] = vf[((long long)(b * TT + t)) * DD + h * HSZ + hs];
}

// ---------------- router: 17 dot products per row (wave per row) ------------
__global__ __launch_bounds__(128) void router_kernel(
    const float* __restrict__ h2, const float* __restrict__ wr, const float* __restrict__ br,
    const float* __restrict__ wn, const float* __restrict__ bn,
    const float* __restrict__ wsk, const float* __restrict__ bsk,
    float* __restrict__ logits, float* __restrict__ nraw, float* __restrict__ skipl) {
  int wid = threadIdx.x >> 5, lane = threadIdx.x & 31;
  int row = blockIdx.x * 4 + wid;
  float aR[EE] = {}, aN[EE] = {}, aS = 0.f;
  for (int d = lane; d < DD; d += 32) {
    float xv = h2[(long long)row * DD + d];
#pragma unroll
    for (int j = 0; j < EE; ++j) {
      aR[j] += xv * wr[d * EE + j];
      aN[j] += xv * wn[d * EE + j];
    }
    aS += xv * wsk[d];
  }
#pragma unroll
  for (int j = 0; j < EE; ++j) {
    for (int off = 16; off; off >>= 1) { aR[j] += __shfl_xor(aR[j], off, 32); aN[j] += __shfl_xor(aN[j], off, 32); }
  }
  for (int off = 16; off; off >>= 1) aS += __shfl_xor(aS, off, 32);
  if (lane == 0) {
#pragma unroll
    for (int j = 0; j < EE; ++j) {
      logits[(long long)row * EE + j] = aR[j] + br[j];
      nraw[(long long)row * EE + j] = aN[j] + bn[j];
    }
    skipl[row] = aS + bsk[0];
  }
}

// ---------------- noisy top-2 + gates + skip ----------------
__global__ void topk_kernel(const float* __restrict__ logits, const float* __restrict__ nraw,
                            const float* __restrict__ noise, const float* __restrict__ skipl,
                            int* __restrict__ idx0, int* __restrict__ idx1,
                            float* __restrict__ gg0, float* __restrict__ gg1,
                            int* __restrict__ nskA, int* __restrict__ n_ns) {
  int t = blockIdx.x * blockDim.x + threadIdx.x;
  if (t >= NT) return;
  float nv[EE];
#pragma unroll
  for (int e = 0; e < EE; ++e) {
    float z = nraw[(long long)t * EE + e];
    float sp = (z > 20.f) ? z : log1pf(__expf(z));
    nv[e] = logits[(long long)t * EE + e] + noise[(long long)t * EE + e] * sp;
  }
  int i0 = 0; float v0 = nv[0];
#pragma unroll
  for (int e = 1; e < EE; ++e) if (nv[e] > v0) { v0 = nv[e]; i0 = e; }
  int i1 = (i0 == 0) ? 1 : 0; float v1 = nv[i1];
#pragma unroll
  for (int e = 0; e < EE; ++e)
    if (e != i0 && nv[e] > v1) { v1 = nv[e]; i1 = e; }
  float g0 = 1.f / (1.f + __expf(v1 - v0));
  idx0[t] = i0; idx1[t] = i1;
  gg0[t] = g0; gg1[t] = 1.f - g0;
  int ns = (skipl[t] <= 0.f) ? 1 : 0;  // nsk = !(sigmoid > 0.5)
  nskA[t] = ns;
  if (ns) atomicAdd(n_ns, 1);
}

__global__ void zero_int(int* p) { if (threadIdx.x == 0) *p = 0; }

// ---------------- per-expert ordered capacity scan ----------------
__global__ __launch_bounds__(256) void expert_scan(
    const int* __restrict__ idx0, const int* __restrict__ idx1, const int* __restrict__ nskA,
    const float* __restrict__ gg0, const float* __restrict__ gg1,
    const int* __restrict__ n_ns, int* __restrict__ tokA, float* __restrict__ gselA) {
  int e = blockIdx.x, tid = threadIdx.x;
  int cap = (*n_ns) * KTOP / EE;
  if (cap > CAPM) cap = CAPM;
  __shared__ int sc[256];
  __shared__ int sRun;
  if (tid == 0) sRun = 0;
  __syncthreads();
  for (int base = 0; base < NT; base += 256) {
    int t = base + tid;
    int m = 0; float g = 0.f;
    if (nskA[t]) {
      if (idx0[t] == e) { m = 1; g = gg0[t]; }
      else if (idx1[t] == e) { m = 1; g = gg1[t]; }
    }
    sc[tid] = m; __syncthreads();
    for (int off = 1; off < 256; off <<= 1) {
      int v = (tid >= off) ? sc[tid - off] : 0;
      __syncthreads();
      sc[tid] += v;
      __syncthreads();
    }
    int rank = sRun + sc[tid] - 1;
    if (m && rank < cap) {
      tokA[(long long)e * CAPM + rank] = t;
      gselA[(long long)e * CAPM + rank] = g;
    }
    __syncthreads();
    if (tid == 0) sRun += sc[255];
    __syncthreads();
  }
  int total = sRun; if (total > cap) total = cap;
  for (int s = total + tid; s < CAPM; s += 256) {
    tokA[(long long)e * CAPM + s] = NT;
    gselA[(long long)e * CAPM + s] = 0.f;
  }
}

// ---------------- out = x1 + (nsk ? 0 : h2)  (experts atomically add later) --
__global__ void out_init(const float* __restrict__ x1, const float* __restrict__ h2,
                         const int* __restrict__ nskA, float* __restrict__ out) {
  long long i = (long long)blockIdx.x * blockDim.x + threadIdx.x;
  if (i >= (long long)NT * DD) return;
  int t = (int)(i / DD);
  out[i] = x1[i] + (nskA[t] ? 0.f : h2[i]);
}

// ---------------- host launcher ----------------
extern "C" void kernel_launch(void* const* d_in, const int* in_sizes, int n_in,
                              void* d_out, int out_size, void* d_ws, size_t ws_size,
                              hipStream_t stream) {
  (void)in_sizes; (void)n_in; (void)out_size; (void)ws_size;
  const float* x     = (const float*)d_in[0];
  const float* noise = (const float*)d_in[1];
  const float* ln1g  = (const float*)d_in[2];
  const float* ln1b  = (const float*)d_in[3];
  const float* wq    = (const float*)d_in[4];
  const float* wk    = (const float*)d_in[5];
  const float* wv    = (const float*)d_in[6];
  const float* wproj = (const float*)d_in[7];
  const float* bproj = (const float*)d_in[8];
  const float* ln2g  = (const float*)d_in[9];
  const float* ln2b  = (const float*)d_in[10];
  const float* wrout = (const float*)d_in[11];
  const float* brout = (const float*)d_in[12];
  const float* wnoi  = (const float*)d_in[13];
  const float* bnoi  = (const float*)d_in[14];
  const float* wskip = (const float*)d_in[15];
  const float* bskip = (const float*)d_in[16];
  const float* ew1   = (const float*)d_in[17];
  const float* eb1   = (const float*)d_in[18];
  const float* ew2   = (const float*)d_in[19];
  const float* eb2   = (const float*)d_in[20];
  float* out = (float*)d_out;

  char* ws = (char*)d_ws;
  size_t off = 0;
  auto alloc = [&](size_t bytes) -> void* {
    void* p = ws + off;
    off = (off + bytes + 255) & ~(size_t)255;
    return p;
  };
  const long long ND = (long long)NT * DD;
  float*    h32    = (float*)alloc(ND * 4);
  _Float16* h16    = (_Float16*)alloc(ND * 2);
  _Float16* wqT    = (_Float16*)alloc((long long)DD * DD * 2);
  _Float16* wkT    = (_Float16*)alloc((long long)DD * DD * 2);
  _Float16* wvT    = (_Float16*)alloc((long long)DD * DD * 2);
  _Float16* wpT    = (_Float16*)alloc((long long)DD * DD * 2);
  _Float16* q16    = (_Float16*)alloc(ND * 2);
  _Float16* k16    = (_Float16*)alloc(ND * 2);
  _Float16* v16b   = (_Float16*)alloc(ND * 2);
  _Float16* vt16   = (_Float16*)alloc(ND * 2);
  _Float16* attn16 = (_Float16*)alloc(ND * 2);
  float*    x1     = (float*)alloc(ND * 4);
  float*    h2     = (float*)alloc(ND * 4);
  _Float16* h2_16  = (_Float16*)alloc(ND * 2);
  _Float16* ew1t   = (_Float16*)alloc((long long)EE * DFFC * DD * 2);
  _Float16* ew2t   = (_Float16*)alloc((long long)EE * DD * DFFC * 2);
  _Float16* mid16  = (_Float16*)alloc((long long)EE * CAPM * DFFC * 2);
  float*    logits = (float*)alloc((long long)NT * EE * 4);
  float*    nraw   = (float*)alloc((long long)NT * EE * 4);
  float*    skipl  = (float*)alloc((long long)NT * 4);
  int*      idx0   = (int*)alloc((long long)NT * 4);
  int*      idx1   = (int*)alloc((long long)NT * 4);
  int*      nskA   = (int*)alloc((long long)NT * 4);
  float*    gg0    = (float*)alloc((long long)NT * 4);
  float*    gg1    = (float*)alloc((long long)NT * 4);
  int*      n_ns   = (int*)alloc(256);
  int*      tokA   = (int*)alloc((long long)EE * CAPM * 4);
  float*    gselA  = (float*)alloc((long long)EE * CAPM * 4);

  // 1) weight convert/transpose
  {
    long long n = (long long)DD * DD;
    dim3 g((unsigned)((n + 255) / 256), 1, 1);
    transpose_cvt<<<g, 256, 0, stream>>>(wq, wqT, DD, DD, 0, 0);
    transpose_cvt<<<g, 256, 0, stream>>>(wk, wkT, DD, DD, 0, 0);
    transpose_cvt<<<g, 256, 0, stream>>>(wv, wvT, DD, DD, 0, 0);
    transpose_cvt<<<g, 256, 0, stream>>>(wproj, wpT, DD, DD, 0, 0);
    long long n1 = (long long)DD * DFFC;
    dim3 ge((unsigned)((n1 + 255) / 256), 1, EE);
    transpose_cvt<<<ge, 256, 0, stream>>>(ew1, ew1t, DD, DFFC, (long long)DD * DFFC, (long long)DFFC * DD);
    transpose_cvt<<<ge, 256, 0, stream>>>(ew2, ew2t, DFFC, DD, (long long)DFFC * DD, (long long)DD * DFFC);
  }
  // 2) LN1
  ln_kernel<<<NT, 256, 0, stream>>>(x, ln1g, ln1b, h32, h16);
  // 3) Q/K/V GEMMs (f16 out); block tile 64x128
  {
    dim3 g(DD / 128, NT / 64, 1);
    gemm_nt<false, false, false, false, false, true, false><<<g, 256, 0, stream>>>(
        h16, DD, 0, wqT, DD, 0, nullptr, 0, nullptr, 0,
        nullptr, 0, q16, 0, DD, nullptr, 0, nullptr, 0, DD, NT);
    gemm_nt<false, false, false, false, false, true, false><<<g, 256, 0, stream>>>(
        h16, DD, 0, wkT, DD, 0, nullptr, 0, nullptr, 0,
        nullptr, 0, k16, 0, DD, nullptr, 0, nullptr, 0, DD, NT);
    gemm_nt<false, false, false, false, false, true, false><<<g, 256, 0, stream>>>(
        h16, DD, 0, wvT, DD, 0, nullptr, 0, nullptr, 0,
        nullptr, 0, v16b, 0, DD, nullptr, 0, nullptr, 0, DD, NT);
  }
  // 4) repack V, flash attention
  pack_vt<<<(unsigned)(ND / 256), 256, 0, stream>>>(v16b, vt16);
  flash_attn<<<dim3(TT / 64, BB * HH), 128, 0, stream>>>(q16, k16, vt16, attn16);
  // 5) x1 = x + attn @ w_proj + b_proj
  {
    dim3 g(DD / 128, NT / 64, 1);
    gemm_nt<false, false, true, true, true, false, false><<<g, 256, 0, stream>>>(
        attn16, DD, 0, wpT, DD, 0, bproj, 0, x, DD,
        x1, 0, nullptr, 0, DD, nullptr, 0, nullptr, 0, DD, NT);
  }
  // 6) LN2, router, top-2, scan
  ln_kernel<<<NT, 256, 0, stream>>>(x1, ln2g, ln2b, h2, h2_16);
  router_kernel<<<NT / 4, 128, 0, stream>>>(h2, wrout, brout, wnoi, bnoi, wskip, bskip,
                                            logits, nraw, skipl);
  zero_int<<<1, 32, 0, stream>>>(n_ns);
  topk_kernel<<<NT / 256, 256, 0, stream>>>(logits, nraw, noise, skipl,
                                            idx0, idx1, gg0, gg1, nskA, n_ns);
  expert_scan<<<EE, 256, 0, stream>>>(idx0, idx1, nskA, gg0, gg1, n_ns, tokA, gselA);
  // 7) out = x1 + (nsk ? 0 : h2); experts atomic-add on top
  out_init<<<(unsigned)(ND / 256), 256, 0, stream>>>(x1, h2, nskA, out);
  // 8) expert FFN: mid = relu(gather(h2) @ ew1 + eb1) -> f16
  {
    dim3 g(DFFC / 128, CAPM / 64, EE);
    gemm_nt<true, true, true, false, false, true, false><<<g, 256, 0, stream>>>(
        h2_16, DD, 0, ew1t, DD, (long long)DFFC * DD, eb1, DFFC, nullptr, 0,
        nullptr, 0, mid16, (long long)CAPM * DFFC, DFFC,
        tokA, CAPM, nullptr, 0, DD, NT);
  }
  // 9) yout = mid @ ew2 + eb2, gate-scaled atomic scatter into out
  {
    dim3 g(DD / 128, CAPM / 64, EE);
    gemm_nt<false, false, true, false, false, false, true><<<g, 256, 0, stream>>>(
        mid16, DFFC, (long long)CAPM * DFFC, ew2t, DFFC, (long long)DD * DFFC, eb2, DD, nullptr, 0,
        out, 0, nullptr, 0, DD,
        tokA, CAPM, gselA, CAPM, DFFC, NT);
  }
}
